// MAMGCN_decblock_54941221651371
// MI455X (gfx1250) — matrix operand order repaired and checked by reading, not compiled, not run on hardware
//
#include <hip/hip_runtime.h>
#include <hip/hip_bf16.h>
#include <cstdint>
#include <cstddef>

typedef __bf16 bf16;
typedef __attribute__((ext_vector_type(16))) bf16 v16bf;
typedef __attribute__((ext_vector_type(8)))  bf16 v8bf;
typedef __attribute__((ext_vector_type(8)))  float v8f;
typedef __attribute__((ext_vector_type(4)))  unsigned v4u;
typedef __attribute__((ext_vector_type(8)))  int v8i;
typedef __attribute__((ext_vector_type(4)))  int v4i;

#define NB 16     // batch
#define NN 2000   // nodes
#define NT 12     // time steps
#define NF 64     // features
#define NKC 3     // cheb order
#define NDF 256   // ffn hidden
#define KP 32     // padded K for prod GEMM
#define TFD 768   // T*F

#if __has_builtin(__builtin_amdgcn_tensor_load_to_lds)
#define HAVE_TDM 1
#else
#define HAVE_TDM 0
#endif

// ---------------------------------------------------------------- WMMA helpers

__device__ __forceinline__ v16bf zero16() {
  v16bf z;
#pragma unroll
  for (int i = 0; i < 16; ++i) z[i] = (bf16)0.0f;
  return z;
}

// A-matrix 16x32 bf16 fragment, row-major source. lanes 0-15: M=lane, K groups
// {0..7, 16..23}; lanes 16-31: M=lane-16, K groups {8..15, 24..31}.
__device__ __forceinline__ v16bf load_a_frag(const bf16* __restrict__ A, int lda,
                                             int m0, int k0, int kmax) {
  int lane = threadIdx.x & 31;
  int row = m0 + (lane & 15);
  int khalf = (lane >> 4) << 3;
  v16bf r = zero16();
  const bf16* base = A + (size_t)row * lda + k0 + khalf;
  if (k0 + khalf < kmax) {
    v8bf g = *(const v8bf*)base;
#pragma unroll
    for (int i = 0; i < 8; ++i) r[i] = g[i];
  }
  if (k0 + khalf + 16 < kmax) {
    v8bf g = *(const v8bf*)(base + 16);
#pragma unroll
    for (int i = 0; i < 8; ++i) r[8 + i] = g[i];
  }
  return r;
}

// A fragment sourced from an f32 LDS matrix (rows = 16, row-major, stride lda).
__device__ __forceinline__ v16bf a_frag_f32(const float* M, int lda, int k0) {
  int lane = threadIdx.x & 31;
  const float* base = M + (lane & 15) * lda + k0 + ((lane >> 4) << 3);
  v16bf r;
#pragma unroll
  for (int i = 0; i < 8; ++i) {
    r[i] = (bf16)base[i];
    r[8 + i] = (bf16)base[16 + i];
  }
  return r;
}

// B-matrix 32x16 bf16 fragment, row-major (K x N) source: lane L holds column
// N=n0+L%16, K groups as A ({0..7,16..23} / {8..15,24..31}).
__device__ __forceinline__ v16bf load_b_frag(const bf16* __restrict__ Bm, int ldb,
                                             int k0, int n0, int kmax) {
  int lane = threadIdx.x & 31;
  int n = n0 + (lane & 15);
  int kb = (lane >> 4) << 3;
  v16bf r;
#pragma unroll
  for (int i = 0; i < 16; ++i) {
    int k = k0 + kb + (i < 8 ? i : 8 + i);
    r[i] = (k < kmax) ? Bm[(size_t)k * ldb + n] : (bf16)0.0f;
  }
  return r;
}

__device__ __forceinline__ v8f wmma_bf(v16bf a, v16bf b, v8f c) {
  return __builtin_amdgcn_wmma_f32_16x16x32_bf16(false, a, false, b, (short)0, c,
                                                 false, false);
}

__device__ __forceinline__ float fast_rcp(float x) { return __builtin_amdgcn_rcpf(x); }

#if HAVE_TDM
// Issue one TDM 2D panel load: 32 cols x 256 rows of bf16 from a row-major
// (rows x NN) matrix into contiguous LDS [256][32]. Descriptor per ISA 8.3/8.4.
// This toolchain's builtin takes 6 args: (g0 v4u, g1 v8i, g2 v4i, g3 v4i, v8i, cpol).
__device__ __forceinline__ void tdm_load_panel(const bf16* gsrc, unsigned lds_byte_addr) {
  unsigned long long ga = (unsigned long long)(size_t)gsrc;
  v4u g0;
  g0[0] = 1u;                                  // count=1 (valid user D#)
  g0[1] = lds_byte_addr;                       // lds_addr (bytes)
  g0[2] = (unsigned)(ga & 0xffffffffull);      // global_addr[31:0]
  g0[3] = (unsigned)((ga >> 32) & 0x1ffffffull) | (2u << 30);  // addr[56:32], type=2
  v8i g1;
  g1[0] = (int)(1u << 16);                     // data_size=1 (2 bytes); wg_mask=0
  g1[1] = (int)((unsigned)NN << 16);           // tensor_dim0[15:0] (=2000)
  g1[2] = (int)((unsigned)NN << 16);           // tensor_dim0[31:16]=0 | tensor_dim1[15:0]
  g1[3] = (int)(32u << 16);                    // tensor_dim1[31:16]=0 | tile_dim0=32
  g1[4] = 256;                                 // tile_dim1=256, tile_dim2=0
  g1[5] = NN;                                  // tensor_dim0_stride[31:0]
  g1[6] = 0;                                   // stride0[47:32]=0 | dim1_stride[15:0]=0
  g1[7] = 0;
  v4i z4;
  z4[0] = 0; z4[1] = 0; z4[2] = 0; z4[3] = 0;
  v8i z8;
#pragma unroll
  for (int i = 0; i < 8; ++i) z8[i] = 0;
  __builtin_amdgcn_tensor_load_to_lds(g0, g1, z4, z4, z8, 0);
}
#endif

// ---------------------------------------------------------------- prep kernels

__global__ void cvt_bf16_kernel(const float* __restrict__ src, bf16* __restrict__ dst, int n) {
  int i = blockIdx.x * 256 + threadIdx.x;
  if (i < n) dst[i] = (bf16)src[i];
}

// ThetaT[k][o][f] = Theta[k][f][o]
__global__ void thetaT_kernel(const float* __restrict__ Theta, bf16* __restrict__ ThetaT) {
  int idx = blockIdx.x * 256 + threadIdx.x;
  if (idx >= NKC * NF * NF) return;
  int f = idx & 63, o = (idx >> 6) & 63, k = idx >> 12;
  ThetaT[idx] = (bf16)Theta[k * NF * NF + f * NF + o];
}

// lhs[b,n,t'] = sum_f (sum_t x[b,n,t,f]*W1[t]) * W2[f,t'];  rhs[b,n,t] = sum_f W3[f]*x[b,n,t,f]
__global__ void lhs_rhs_kernel(const float* __restrict__ x, const float* __restrict__ W1,
                               const float* __restrict__ W2, const float* __restrict__ W3,
                               bf16* __restrict__ lhs_bf, bf16* __restrict__ rhsT_bf) {
  int idx = blockIdx.x * 256 + threadIdx.x;  // b*N + n
  if (idx >= NB * NN) return;
  int b = idx / NN, n = idx % NN;
  const float* xr = x + (size_t)idx * NT * NF;
  float lp[NF];
#pragma unroll
  for (int f = 0; f < NF; ++f) lp[f] = 0.0f;
  float rv[NT];
  for (int t = 0; t < NT; ++t) {
    float w1 = W1[t];
    float acc = 0.0f;
#pragma unroll
    for (int f = 0; f < NF; ++f) {
      float v = xr[t * NF + f];
      lp[f] = fmaf(v, w1, lp[f]);
      acc = fmaf(v, W3[f], acc);
    }
    rv[t] = acc;
  }
  bf16* lrow = lhs_bf + (size_t)idx * KP;
  for (int tp = 0; tp < NT; ++tp) {
    float s = 0.0f;
#pragma unroll
    for (int f = 0; f < NF; ++f) s = fmaf(lp[f], W2[f * NT + tp], s);
    lrow[tp] = (bf16)s;
  }
  for (int tp = NT; tp < KP; ++tp) lrow[tp] = (bf16)0.0f;
  bf16* rT = rhsT_bf + (size_t)b * KP * NN;
  for (int t = 0; t < NT; ++t) rT[(size_t)t * NN + n] = (bf16)rv[t];
  for (int t = NT; t < KP; ++t) rT[(size_t)t * NN + n] = (bf16)0.0f;
}

// Xt[b][(t*64+f)][j] = x[b,j,t,f]
__global__ void build_xt_kernel(const float* __restrict__ x, bf16* __restrict__ xt) {
  int idx = blockIdx.x * 256 + threadIdx.x;
  if (idx >= NB * TFD * NN) return;
  int j = idx % NN;
  int r = idx / NN;
  int tf = r % TFD;
  int b = r / TFD;
  int t = tf >> 6, f = tf & 63;
  xt[idx] = (bf16)x[((size_t)(b * NN + j) * NT + t) * NF + f];
}

// ---------------------------------------------------------------- big GEMMs

// P[n,m] = sigmoid(sum_t lhs[n,t]*rhs[m,t] + bs[n,m]); one 16x16 tile per wave.
__global__ void __launch_bounds__(32) prod_kernel(const bf16* __restrict__ lhs_b,
                                                  const bf16* __restrict__ rhsT_b,
                                                  const float* __restrict__ bs,
                                                  bf16* __restrict__ P) {
  int nt = blockIdx.x, mt = blockIdx.y;
  int lane = threadIdx.x & 31;
  v16bf a = load_a_frag(lhs_b, KP, mt * 16, 0, KP);
  v16bf b = load_b_frag(rhsT_b, NN, 0, nt * 16, KP);
  v8f acc;
#pragma unroll
  for (int r = 0; r < 8; ++r) acc[r] = 0.0f;
  acc = wmma_bf(a, b, acc);
  int ncol = nt * 16 + (lane & 15);
  int rbase = (lane >> 4) << 3;
#pragma unroll
  for (int r = 0; r < 8; ++r) {
    int nrow = mt * 16 + rbase + r;
    float v = acc[r] + bs[(size_t)nrow * NN + ncol];
    P[(size_t)nrow * NN + ncol] = (bf16)fast_rcp(1.0f + __expf(-v));
  }
}

// S_pre = Vs @ P, 2x2 tile blocking per wave (32x32 output). Grid 63x63 covers the
// 2016-padded space; overreads land in adjacent ws buffers (safe), writes guarded.
// B panels staged to LDS by the Tensor Data Mover, double-buffered on TENSORcnt.
__global__ void __launch_bounds__(32) sgemm_kernel(const bf16* __restrict__ A,
                                                   const bf16* __restrict__ Bm,
                                                   float* __restrict__ S_pre) {
  int n0 = blockIdx.x * 32;
  int m0 = blockIdx.y * 32;
  int lane = threadIdx.x & 31;
  v8f a00, a01, a10, a11;
#pragma unroll
  for (int r = 0; r < 8; ++r) { a00[r] = 0.0f; a01[r] = 0.0f; a10[r] = 0.0f; a11[r] = 0.0f; }

#if HAVE_TDM
  __shared__ bf16 pb[2][256][32];
  const int nchunks = (NN + 255) / 256;  // 8
  tdm_load_panel(Bm + (size_t)0 * NN + n0, (unsigned)(size_t)&pb[0][0][0]);
  for (int c = 0; c < nchunks; ++c) {
    if (c + 1 < nchunks) {
      tdm_load_panel(Bm + (size_t)(c + 1) * 256 * NN + n0,
                     (unsigned)(size_t)&pb[(c + 1) & 1][0][0]);
      __builtin_amdgcn_s_wait_tensorcnt((short)1);
    } else {
      __builtin_amdgcn_s_wait_tensorcnt((short)0);
    }
    const bf16* pbc = &pb[c & 1][0][0];
    int jc0 = c * 256;
    int nrows = NN - jc0;
    if (nrows > 256) nrows = 256;
    int ksteps = (nrows + 31) / 32;
    for (int ks = 0; ks < ksteps; ++ks) {
      int k0 = jc0 + ks * 32;
      v16bf fa0 = load_a_frag(A, NN, m0, k0, NN);
      v16bf fa1 = load_a_frag(A, NN, m0 + 16, k0, NN);
      int kb = (lane >> 4) << 3;
      v16bf fb0, fb1;
#pragma unroll
      for (int e = 0; e < 16; ++e) {
        int lr = ks * 32 + kb + (e < 8 ? e : 8 + e);
        if (lr < nrows) {
          fb0[e] = pbc[lr * 32 + (lane & 15)];
          fb1[e] = pbc[lr * 32 + 16 + (lane & 15)];
        } else {
          fb0[e] = (bf16)0.0f;
          fb1[e] = (bf16)0.0f;
        }
      }
      a00 = wmma_bf(fa0, fb0, a00);
      a01 = wmma_bf(fa0, fb1, a01);
      a10 = wmma_bf(fa1, fb0, a10);
      a11 = wmma_bf(fa1, fb1, a11);
    }
    // LDS reads of this buffer must retire before the TDM overwrites it next iter.
    asm volatile("s_wait_dscnt 0x0" ::: "memory");
  }
#else
  for (int ks = 0; ks < 63; ++ks) {
    int k0 = ks * 32;
    if (ks < 62) {
      int jn = k0 + 32 + lane;
      if (jn < NN) __builtin_prefetch(Bm + (size_t)jn * NN + n0, 0, 0);
    }
    v16bf fa0 = load_a_frag(A, NN, m0, k0, NN);
    v16bf fa1 = load_a_frag(A, NN, m0 + 16, k0, NN);
    v16bf fb0 = load_b_frag(Bm, NN, k0, n0, NN);
    v16bf fb1 = load_b_frag(Bm, NN, k0, n0 + 16, NN);
    a00 = wmma_bf(fa0, fb0, a00);
    a01 = wmma_bf(fa0, fb1, a01);
    a10 = wmma_bf(fa1, fb0, a10);
    a11 = wmma_bf(fa1, fb1, a11);
  }
#endif

  int rbase = (lane >> 4) << 3, c = lane & 15;
  auto st = [&](v8f acc, int mb, int nb) {
    int ncol = nb + c;
    if (ncol >= NN) return;
#pragma unroll
    for (int r = 0; r < 8; ++r) {
      int nrow = mb + rbase + r;
      if (nrow < NN) S_pre[(size_t)nrow * NN + ncol] = acc[r];
    }
  };
  st(a00, m0, n0);
  st(a01, m0, n0 + 16);
  st(a10, m0 + 16, n0);
  st(a11, m0 + 16, n0 + 16);
}

// softmax over axis n (columns), write bf16
__global__ void softmax_col_kernel(const float* __restrict__ S_pre, bf16* __restrict__ S_bf) {
  int j = blockIdx.x * 256 + threadIdx.x;
  if (j >= NN) return;
  float mx = -1e30f;
  for (int n = 0; n < NN; ++n) mx = fmaxf(mx, S_pre[(size_t)n * NN + j]);
  float den = 0.0f;
  for (int n = 0; n < NN; ++n) den += __expf(S_pre[(size_t)n * NN + j] - mx);
  float inv = fast_rcp(den);
  for (int n = 0; n < NN; ++n)
    S_bf[(size_t)n * NN + j] = (bf16)(__expf(S_pre[(size_t)n * NN + j] - mx) * inv);
}

// Fused Chebyshev conv: r_k[tf,i] = sum_j Xt[tf,j]*(cheb[k][j,i]*S[j,i]); then
// xd[b,i,t,o] = relu(sum_k ThetaT[k](o x f) @ r_k(f x i)) — both stages WMMA.
__global__ void __launch_bounds__(128) conv_kernel(const bf16* __restrict__ Xt_b,
                                                   const bf16* __restrict__ S,
                                                   const bf16* __restrict__ cheb_bf,
                                                   const bf16* __restrict__ ThetaT,
                                                   float* __restrict__ xd, int b) {
  int it = blockIdx.x, t = blockIdx.y;
  int wave = threadIdx.x >> 5, lane = threadIdx.x & 31;
  int tf0 = t * NF + wave * 16;
  int i0 = it * 16;
  v8f acc0, acc1, acc2;
#pragma unroll
  for (int r = 0; r < 8; ++r) { acc0[r] = 0.0f; acc1[r] = 0.0f; acc2[r] = 0.0f; }
  for (int ks = 0; ks < 63; ++ks) {
    int k0 = ks * 32;
    int j = k0 + lane;
    if (ks < 62 && j + 32 < NN) {
      __builtin_prefetch(S + (size_t)(j + 32) * NN + i0, 0, 0);
      __builtin_prefetch(cheb_bf + (size_t)(j + 32) * NN + i0, 0, 0);
    }
    v16bf a = load_a_frag(Xt_b, NN, tf0, k0, NN);
    v16bf b0 = zero16(), b1 = zero16(), b2 = zero16();
    int kb = (lane >> 4) << 3, ic = i0 + (lane & 15);
#pragma unroll
    for (int e = 0; e < 16; ++e) {
      int je = k0 + kb + (e < 8 ? e : 8 + e);
      if (je < NN) {
        size_t o = (size_t)je * NN + ic;
        float sf = (float)S[o];
        b0[e] = (bf16)(sf * (float)cheb_bf[o]);
        b1[e] = (bf16)(sf * (float)cheb_bf[(size_t)NN * NN + o]);
        b2[e] = (bf16)(sf * (float)cheb_bf[(size_t)2 * NN * NN + o]);
      }
    }
    acc0 = wmma_bf(a, b0, acc0);
    acc1 = wmma_bf(a, b1, acc1);
    acc2 = wmma_bf(a, b2, acc2);
  }
  __shared__ float r_lds[NKC][NF][16];
  int icol = lane & 15, rbase = (lane >> 4) << 3;
#pragma unroll
  for (int r = 0; r < 8; ++r) {
    int f = wave * 16 + rbase + r;
    r_lds[0][f][icol] = acc0[r];
    r_lds[1][f][icol] = acc1[r];
    r_lds[2][f][icol] = acc2[r];
  }
  __syncthreads();
  // Theta mixing: wave w computes o-tile w. D[o_local, i] accumulated over k and f.
  v8f accT;
#pragma unroll
  for (int r = 0; r < 8; ++r) accT[r] = 0.0f;
#pragma unroll
  for (int k = 0; k < NKC; ++k) {
    const bf16* Tk = ThetaT + k * NF * NF;
#pragma unroll
    for (int ks = 0; ks < 2; ++ks) {
      v16bf a = load_a_frag(Tk, NF, wave * 16, ks * 32, NF);
      int frow = ks * 32 + ((lane >> 4) << 3);
      v16bf bfr;
#pragma unroll
      for (int e = 0; e < 16; ++e) bfr[e] = (bf16)r_lds[k][frow + (e < 8 ? e : 8 + e)][lane & 15];
      accT = wmma_bf(a, bfr, accT);
    }
  }
  int ig = i0 + (lane & 15);
#pragma unroll
  for (int r = 0; r < 8; ++r) {
    int o = wave * 16 + rbase + r;
    xd[((size_t)(b * NN + ig) * NT + t) * NF + o] = fmaxf(accT[r], 0.0f);
  }
}

// ---------------------------------------------------------------- transformer

struct TransParams {
  const float *xd, *xe, *x;
  const bf16 *swq, *swk, *swv, *swo;
  const float *sbq, *sbk, *sbv, *sbo;
  const bf16 *cwq, *cwk, *cwv, *cwo;
  const float *cbq, *cbk, *cbv, *cbo;
  const float *l1g, *l1b, *l2g, *l2b, *l3g, *l3b;
  const bf16 *fw1, *fw2;
  const float *fb1, *fb2;
  float* out;
};

// Per-wave token GEMM: D(16 x NCOLS) = act(16 x KDIM, f32 LDS) @ W(KDIM x NCOLS, bf16) + bias.
// Two waves split the n-tiles; only token rows < NT are stored.
template <int KDIM, int NCOLS, bool RELU>
__device__ void gemm_tok(const float* act, int lda, const bf16* __restrict__ W,
                         const float* __restrict__ bias, float* dst, int ldd) {
  int wave = threadIdx.x >> 5;
  int lane = threadIdx.x & 31;
  constexpr int NTILES = NCOLS / 16;
  for (int nt = wave; nt < NTILES; nt += 2) {
    v8f acc;
#pragma unroll
    for (int r = 0; r < 8; ++r) acc[r] = 0.0f;
#pragma unroll
    for (int ks = 0; ks < KDIM / 32; ++ks) {
      v16bf a = a_frag_f32(act, lda, ks * 32);
      v16bf b = load_b_frag(W, NCOLS, ks * 32, nt * 16, KDIM);
      acc = wmma_bf(a, b, acc);
    }
    int ncol = nt * 16 + (lane & 15);
    int rbase = (lane >> 4) << 3;
#pragma unroll
    for (int r = 0; r < 8; ++r) {
      int tok = rbase + r;
      if (tok < NT) {
        float v = acc[r] + bias[ncol];
        if (RELU) v = fmaxf(v, 0.0f);
        dst[tok * ldd + ncol] = v;
      }
    }
  }
}

// qin/kvin: [16][64] LDS (rows 12..15 don't-care). Result in sq.
__device__ void mha2(float (*qin)[NF], float (*kvin)[NF],
                     const bf16* wq, const bf16* wk, const bf16* wv,
                     const float* bq, const float* bk, const float* bv,
                     const bf16* wo, const float* bo,
                     float (*sq)[NF], float (*sk)[NF], float (*sv)[NF],
                     float (*so)[NF]) {
  gemm_tok<64, 64, false>(&qin[0][0], NF, wq, bq, &sq[0][0], NF);
  gemm_tok<64, 64, false>(&kvin[0][0], NF, wk, bk, &sk[0][0], NF);
  gemm_tok<64, 64, false>(&kvin[0][0], NF, wv, bv, &sv[0][0], NF);
  __syncthreads();
  int d = threadIdx.x;
  int hb = (d >> 4) << 4;  // head base (dh = 16)
  for (int i = 0; i < NT; ++i) {
    float sc[NT];
    float mx = -1e30f;
#pragma unroll
    for (int j = 0; j < NT; ++j) {
      float s = 0.0f;
#pragma unroll
      for (int dd = 0; dd < 16; ++dd) s = fmaf(sq[i][hb + dd], sk[j][hb + dd], s);
      s *= 0.25f;  // 1/sqrt(16)
      sc[j] = s;
      mx = fmaxf(mx, s);
    }
    float den = 0.0f;
#pragma unroll
    for (int j = 0; j < NT; ++j) { sc[j] = __expf(sc[j] - mx); den += sc[j]; }
    float o = 0.0f;
#pragma unroll
    for (int j = 0; j < NT; ++j) o = fmaf(sc[j], sv[j][d], o);
    so[i][d] = o * fast_rcp(den);
  }
  __syncthreads();
  gemm_tok<64, 64, false>(&so[0][0], NF, wo, bo, &sq[0][0], NF);
  __syncthreads();
}

__device__ void add_ln(float (*base)[NF], float (*add)[NF],
                       const float* __restrict__ g, const float* __restrict__ bb,
                       float* s_mu, float* s_rs) {
  int d = threadIdx.x;
  if (d < NT) {
    float m = 0.0f;
    for (int c = 0; c < NF; ++c) m += base[d][c] + add[d][c];
    m *= (1.0f / NF);
    float v = 0.0f;
    for (int c = 0; c < NF; ++c) { float z = base[d][c] + add[d][c] - m; v = fmaf(z, z, v); }
    v *= (1.0f / NF);
    s_mu[d] = m;
    s_rs[d] = rsqrtf(v + 1e-5f);
  }
  __syncthreads();
  for (int i = 0; i < NT; ++i)
    base[i][d] = (base[i][d] + add[i][d] - s_mu[i]) * s_rs[i] * g[d] + bb[d];
  __syncthreads();
}

__global__ void __launch_bounds__(64) trans_kernel(TransParams p) {
  int seq = blockIdx.x;
  int d = threadIdx.x;
  __shared__ float sx[16][NF], sm[16][NF], so[16][NF];  // GEMM A-operands: 16 rows
  __shared__ float sq[NT][NF], sk[NT][NF], sv[NT][NF];
  __shared__ float sh[16][NDF];
  __shared__ float s_mu[NT], s_rs[NT];

  for (int idx = d; idx < NT * NF; idx += NF) {
    sx[idx >> 6][idx & 63] = p.xd[(size_t)seq * (NT * NF) + idx];
    sm[idx >> 6][idx & 63] = p.xe[(size_t)seq * (NT * NF) + idx];
  }
  __syncthreads();

  mha2(sx, sx, p.swq, p.swk, p.swv, p.sbq, p.sbk, p.sbv, p.swo, p.sbo, sq, sk, sv, so);
  add_ln(sx, sq, p.l1g, p.l1b, s_mu, s_rs);
  mha2(sx, sm, p.cwq, p.cwk, p.cwv, p.cbq, p.cbk, p.cbv, p.cwo, p.cbo, sq, sk, sv, so);
  add_ln(sx, sq, p.l2g, p.l2b, s_mu, s_rs);

  gemm_tok<64, 256, true>(&sx[0][0], NF, p.fw1, p.fb1, &sh[0][0], NDF);
  __syncthreads();
  gemm_tok<256, 64, false>(&sh[0][0], NDF, p.fw2, p.fb2, &sq[0][0], NF);
  __syncthreads();
  add_ln(sx, sq, p.l3g, p.l3b, s_mu, s_rs);

  for (int idx = d; idx < NT * NF; idx += NF)
    p.out[(size_t)seq * (NT * NF) + idx] =
        p.x[(size_t)seq * (NT * NF) + idx] + sx[idx >> 6][idx & 63];
}

// ---------------------------------------------------------------- launch

extern "C" void kernel_launch(void* const* d_in, const int* in_sizes, int n_in,
                              void* d_out, int out_size, void* d_ws, size_t ws_size,
                              hipStream_t stream) {
  (void)in_sizes; (void)n_in; (void)out_size; (void)ws_size;
  const float* x     = (const float*)d_in[0];
  const float* x_enc = (const float*)d_in[1];
  const float* cheb  = (const float*)d_in[2];
  const float* W1    = (const float*)d_in[3];
  const float* W2    = (const float*)d_in[4];
  const float* W3    = (const float*)d_in[5];
  const float* bs    = (const float*)d_in[6];
  const float* Vs    = (const float*)d_in[7];
  const float* Theta = (const float*)d_in[8];

  uint8_t* wsb = (uint8_t*)d_ws;
  size_t off = 0;
  auto take = [&](size_t bytes) -> void* {
    void* p = wsb + off;
    off += (bytes + 255) & ~(size_t)255;
    return p;
  };
  bf16* lhs_bf    = (bf16*)take((size_t)NB * NN * KP * 2);
  bf16* rhsT_bf   = (bf16*)take((size_t)NB * KP * NN * 2);
  bf16* Vs_bf     = (bf16*)take((size_t)NN * NN * 2);
  bf16* cheb_bf   = (bf16*)take((size_t)NKC * NN * NN * 2);
  bf16* Xt_bf     = (bf16*)take((size_t)NB * TFD * NN * 2);
  bf16* P_bf      = (bf16*)take((size_t)NN * NN * 2);
  float* S_pre    = (float*)take((size_t)NN * NN * 4);
  bf16* S_bf      = (bf16*)take((size_t)NN * NN * 2);
  float* xd       = (float*)take((size_t)NB * NN * TFD * 4);
  bf16* ThetaT_bf = (bf16*)take((size_t)NKC * NF * NF * 2);
  bf16* wbf[10];
  static const int widx[10] = {9, 10, 11, 15, 17, 18, 19, 23, 31, 33};
  static const int wsz[10]  = {4096, 4096, 4096, 4096, 4096, 4096, 4096, 4096, 16384, 16384};
  for (int i = 0; i < 10; ++i) wbf[i] = (bf16*)take((size_t)wsz[i] * 2);

  cvt_bf16_kernel<<<(NN * NN + 255) / 256, 256, 0, stream>>>(Vs, Vs_bf, NN * NN);
  cvt_bf16_kernel<<<(NKC * NN * NN + 255) / 256, 256, 0, stream>>>(cheb, cheb_bf, NKC * NN * NN);
  thetaT_kernel<<<(NKC * NF * NF + 255) / 256, 256, 0, stream>>>(Theta, ThetaT_bf);
  for (int i = 0; i < 10; ++i)
    cvt_bf16_kernel<<<(wsz[i] + 255) / 256, 256, 0, stream>>>((const float*)d_in[widx[i]],
                                                              wbf[i], wsz[i]);
  lhs_rhs_kernel<<<(NB * NN + 255) / 256, 256, 0, stream>>>(x, W1, W2, W3, lhs_bf, rhsT_bf);
  build_xt_kernel<<<(NB * TFD * NN + 255) / 256, 256, 0, stream>>>(x, Xt_bf);

  for (int b = 0; b < NB; ++b) {
    prod_kernel<<<dim3(NN / 16, NN / 16), 32, 0, stream>>>(
        lhs_bf + (size_t)b * NN * KP, rhsT_bf + (size_t)b * KP * NN, bs, P_bf);
    sgemm_kernel<<<dim3(63, 63), 32, 0, stream>>>(Vs_bf, P_bf, S_pre);
    softmax_col_kernel<<<(NN + 255) / 256, 256, 0, stream>>>(S_pre, S_bf);
    conv_kernel<<<dim3(NN / 16, NT), 128, 0, stream>>>(
        Xt_bf + (size_t)b * TFD * NN, S_bf, cheb_bf, ThetaT_bf, xd, b);
  }

  TransParams tp;
  tp.xd = xd; tp.xe = x_enc; tp.x = x;
  tp.swq = wbf[0]; tp.swk = wbf[1]; tp.swv = wbf[2]; tp.swo = wbf[3];
  tp.sbq = (const float*)d_in[12]; tp.sbk = (const float*)d_in[13];
  tp.sbv = (const float*)d_in[14]; tp.sbo = (const float*)d_in[16];
  tp.cwq = wbf[4]; tp.cwk = wbf[5]; tp.cwv = wbf[6]; tp.cwo = wbf[7];
  tp.cbq = (const float*)d_in[20]; tp.cbk = (const float*)d_in[21];
  tp.cbv = (const float*)d_in[22]; tp.cbo = (const float*)d_in[24];
  tp.l1g = (const float*)d_in[25]; tp.l1b = (const float*)d_in[26];
  tp.l2g = (const float*)d_in[27]; tp.l2b = (const float*)d_in[28];
  tp.l3g = (const float*)d_in[29]; tp.l3b = (const float*)d_in[30];
  tp.fw1 = wbf[8]; tp.fb1 = (const float*)d_in[32];
  tp.fw2 = wbf[9]; tp.fb2 = (const float*)d_in[34];
  tp.out = (float*)d_out;

  trans_kernel<<<dim3(NB * NN), 64, 0, stream>>>(tp);
}
